// GroupNet_38852274160107
// MI455X (gfx1250) — matrix operand order repaired and checked by reading, not compiled, not run on hardware
//
#include <hip/hip_runtime.h>
#include <hip/hip_bf16.h>
#include <stdint.h>

#define Bsz 16384
#define Nn  64
#define Hh  128
#define IGc 8
#define ODc 28
#define SDc 36
#define ROWLEN (SDc + Nn * ODc)   // 1828

typedef __attribute__((ext_vector_type(16))) __bf16 v16bf;
typedef __attribute__((ext_vector_type(8)))  float  v8f;

__device__ __forceinline__ uint16_t f2bf(float f) {
    uint32_t x = __float_as_uint(f);
    uint32_t r = x + 0x7FFFu + ((x >> 16) & 1u);   // round-to-nearest-even
    return (uint16_t)(r >> 16);
}
__device__ __forceinline__ float bf2f(uint16_t b) {
    return __uint_as_float(((uint32_t)b) << 16);
}
__device__ __forceinline__ float fast_tanh(float f) {
#if __has_builtin(__builtin_amdgcn_tanhf)
    return __builtin_amdgcn_tanhf(f);     // V_TANH_F32 (gfx1250 trans op)
#else
    return tanhf(f);
#endif
}

// ---------------------------------------------------------------------------
// Prep: swizzle We (28x128, K-padded to 32) and Ww (256x128) into bf16
// B-fragment layout in d_ws.
//   B frag (bf16, KxN=32x16, 8 VGPRs): lane = h*16+n ; VGPR v packs K=16h+2v, 16h+2v+1
//   ws[0      .. 2048)  : We_frag  [ntile=8][lane=32][v=8]  (u32, 2x bf16)
//   ws[2048 .. 18432)   : Ww_frag  [ntile=8][kstep=8][lane=32][v=8]
// ---------------------------------------------------------------------------
__global__ void prep_weights(const float* __restrict__ We,
                             const float* __restrict__ Ww,
                             uint32_t* __restrict__ ws) {
    int t = blockIdx.x * blockDim.x + threadIdx.x;
    if (t < 2048) {                       // We fragments
        int v = t & 7, lane = (t >> 3) & 31, ntile = t >> 8;
        int h = lane >> 4, col = ntile * 16 + (lane & 15);
        int K = 16 * h + 2 * v;
        float lo = (K     < ODc) ? We[K * Hh + col]       : 0.0f;
        float hi = (K + 1 < ODc) ? We[(K + 1) * Hh + col] : 0.0f;
        ws[t] = ((uint32_t)f2bf(hi) << 16) | f2bf(lo);
    } else if (t < 2048 + 16384) {        // Ww fragments
        int u = t - 2048;
        int v = u & 7, lane = (u >> 3) & 31, kstep = (u >> 8) & 7, ntile = u >> 11;
        int h = lane >> 4, col = ntile * 16 + (lane & 15);
        int K = kstep * 32 + 16 * h + 2 * v;
        float lo = Ww[K * Hh + col];
        float hi = Ww[(K + 1) * Hh + col];
        ws[t] = ((uint32_t)f2bf(hi) << 16) | f2bf(lo);
    }
}

// ---------------------------------------------------------------------------
// Main: one block (8 wave32) per batch row.
// ---------------------------------------------------------------------------
__global__ __launch_bounds__(256) void groupnet_kernel(
    const float* __restrict__ x,
    const float* __restrict__ Ws, const float* __restrict__ bs,
    const float* __restrict__ be, const float* __restrict__ bw,
    const float* __restrict__ vv,
    const float* __restrict__ Wi, const float* __restrict__ bi,
    const float* __restrict__ Wa, const float* __restrict__ ba,
    const uint32_t* __restrict__ wfrag,
    float* __restrict__ out)
{
    __shared__ float    other_f[Nn * ODc];     // raw entities, f32     (7 KB)
    __shared__ uint16_t other_b[Nn * 32];      // bf16, K padded        (4 KB)
    __shared__ uint16_t attin[Nn * 256];       // [other_es | self_e]   (32 KB)
    __shared__ float    atts[Nn * Hh];         // tanh(att)             (32 KB)
    __shared__ float    selfe[Hh];
    __shared__ uint32_t selfe_b[Hh / 2];       // packed bf16 pairs
    __shared__ float    scores[Nn];
    __shared__ float    wts[Nn];
    __shared__ int      iidx[IGc];
    __shared__ int      used[Nn];
    __shared__ float    ivec[IGc * ODc];
    __shared__ float    ue[Hh], ie[Hh];

    const int row = blockIdx.x;
    const int tid = threadIdx.x;
    const float* xr = x + (size_t)row * ROWLEN;

    // ---- load entity block, build padded bf16 copy --------------------------
    for (int i = tid; i < Nn * ODc; i += 256) other_f[i] = xr[SDc + i];
    __syncthreads();
    for (int i = tid; i < Nn * 32; i += 256) {
        int n = i >> 5, k = i & 31;
        other_b[i] = (k < ODc) ? f2bf(other_f[n * ODc + k]) : (uint16_t)0;
    }
    // ---- self_e = relu(self_info @ Ws + bs) --------------------------------
    if (tid < Hh) {
        float a = bs[tid];
        #pragma unroll 4
        for (int k = 0; k < SDc; ++k) a = fmaf(xr[k], Ws[k * Hh + tid], a);
        selfe[tid] = fmaxf(a, 0.0f);
    }
    __syncthreads();
    if (tid < Hh / 2) {
        selfe_b[tid] = ((uint32_t)f2bf(selfe[2 * tid + 1]) << 16) | f2bf(selfe[2 * tid]);
    }
    __syncthreads();
    // broadcast self_e (packed bf16) into attin columns 128..255
    {
        uint32_t* attin32 = (uint32_t*)attin;
        for (int i = tid; i < Nn * (Hh / 2); i += 256) {
            int n = i >> 6, hp = i & 63;
            attin32[n * 128 + 64 + hp] = selfe_b[hp];
        }
    }
    __syncthreads();

    const int wave  = tid >> 5;
    const int lane  = tid & 31;
    const int hhalf = lane >> 4;
    const int lcol  = lane & 15;
    const int ntile = wave;                 // each wave owns a 16-col N tile

    // ---- GEMM1: other(64x32 bf16) @ We -> relu -> attin[:, :128] -----------
    {
        union { uint32_t u[8]; v16bf v; } bfr;
        const uint32_t* bp = wfrag + ((size_t)ntile * 32 + lane) * 8;
        #pragma unroll
        for (int q = 0; q < 8; ++q) bfr.u[q] = bp[q];
        #pragma unroll
        for (int mt = 0; mt < 4; ++mt) {
            union { uint32_t u[8]; v16bf v; } afr;
            int m = mt * 16 + lcol;
            #pragma unroll
            for (int q = 0; q < 8; ++q) {
                int K = 16 * hhalf + 2 * q;
                afr.u[q] = *(const uint32_t*)&other_b[m * 32 + K];
            }
            v8f acc = {};
            acc = __builtin_amdgcn_wmma_f32_16x16x32_bf16(
                      false, afr.v, false, bfr.v, (short)0, acc, false, false);
            int col = ntile * 16 + lcol;
            float bec = be[col];
            #pragma unroll
            for (int r = 0; r < 8; ++r) {
                int mm = mt * 16 + 8 * hhalf + r;
                attin[mm * 256 + col] = f2bf(fmaxf(acc[r] + bec, 0.0f));
            }
        }
    }
    __syncthreads();

    // ---- GEMM2: attin(64x256 bf16) @ Ww -> tanh -> atts (f32) --------------
    {
        // Hoist all 8 K-step B fragments (L2-resident) into registers once.
        union { uint32_t u[8]; v16bf v; } bfr[8];
        const uint32_t* wwbase = wfrag + 2048 + (size_t)ntile * 2048 + lane * 8;
        #pragma unroll
        for (int ks = 0; ks < 8; ++ks) {
            const uint32_t* bp = wwbase + (size_t)ks * 256;
            #pragma unroll
            for (int q = 0; q < 8; ++q) bfr[ks].u[q] = bp[q];
        }
        #pragma unroll
        for (int mt = 0; mt < 4; ++mt) {
            v8f acc = {};
            int m = mt * 16 + lcol;
            #pragma unroll
            for (int ks = 0; ks < 8; ++ks) {
                union { uint32_t u[8]; v16bf v; } afr;
                #pragma unroll
                for (int q = 0; q < 8; ++q) {
                    int K = ks * 32 + 16 * hhalf + 2 * q;
                    afr.u[q] = *(const uint32_t*)&attin[m * 256 + K];
                }
                acc = __builtin_amdgcn_wmma_f32_16x16x32_bf16(
                          false, afr.v, false, bfr[ks].v, (short)0, acc, false, false);
            }
            int col = ntile * 16 + lcol;
            float bwc = bw[col];
            #pragma unroll
            for (int r = 0; r < 8; ++r) {
                int mm = mt * 16 + 8 * hhalf + r;
                atts[mm * Hh + col] = fast_tanh(acc[r] + bwc);
            }
        }
    }
    __syncthreads();

    // ---- scores = att @ v ---------------------------------------------------
    if (tid < Nn) {
        float s = 0.0f;
        #pragma unroll 4
        for (int h = 0; h < Hh; ++h) s = fmaf(atts[tid * Hh + h], vv[h], s);
        scores[tid] = s;
    }
    __syncthreads();

    // ---- softmax + ordered top-8 selection (serial; tiny) -------------------
    if (tid == 0) {
        float mx = -1e30f;
        for (int n = 0; n < Nn; ++n) mx = fmaxf(mx, scores[n]);
        float sum = 0.0f;
        for (int n = 0; n < Nn; ++n) { float e = expf(scores[n] - mx); wts[n] = e; sum += e; }
        float inv = 1.0f / sum;
        for (int n = 0; n < Nn; ++n) { wts[n] *= inv; used[n] = 0; }
        for (int j = 0; j < IGc; ++j) {
            int   bidx = 0;  float bval = -1.0f;
            for (int n = 0; n < Nn; ++n)
                if (!used[n] && wts[n] > bval) { bval = wts[n]; bidx = n; }
            used[bidx] = 1;
            iidx[j] = bidx;
        }
    }
    __syncthreads();

    // ---- weights output -----------------------------------------------------
    if (tid < Nn) out[(size_t)Bsz * Hh + (size_t)row * Nn + tid] = wts[tid];

    // ---- i-group vector (weighted raw features, flattened 8x28) -------------
    if (tid < IGc * ODc) {
        int j = tid / ODc, k = tid - j * ODc;
        int n = iidx[j];
        ivec[tid] = other_f[n * ODc + k] * wts[n];
    }
    // ---- u_e = sum over non-top entities of w[n]*other_es[n] ----------------
    if (tid < Hh) {
        float a = 0.0f;
        for (int n = 0; n < Nn; ++n) {
            if (used[n]) continue;
            a = fmaf(wts[n], bf2f(attin[n * 256 + tid]), a);
        }
        ue[tid] = a;
    }
    __syncthreads();

    // ---- i_e = relu(ivec @ Wi + bi) -----------------------------------------
    if (tid < Hh) {
        float a = bi[tid];
        #pragma unroll 4
        for (int k = 0; k < IGc * ODc; ++k) a = fmaf(ivec[k], Wi[k * Hh + tid], a);
        ie[tid] = fmaxf(a, 0.0f);
    }
    __syncthreads();

    // ---- obs_e = concat(self_e, i_e, u_e) @ Wa + ba -------------------------
    if (tid < Hh) {
        float a = ba[tid];
        #pragma unroll 4
        for (int k = 0; k < Hh; ++k) a = fmaf(selfe[k], Wa[k * Hh + tid], a);
        #pragma unroll 4
        for (int k = 0; k < Hh; ++k) a = fmaf(ie[k], Wa[(Hh + k) * Hh + tid], a);
        #pragma unroll 4
        for (int k = 0; k < Hh; ++k) a = fmaf(ue[k], Wa[(2 * Hh + k) * Hh + tid], a);
        out[(size_t)row * Hh + tid] = a;
    }
}

extern "C" void kernel_launch(void* const* d_in, const int* in_sizes, int n_in,
                              void* d_out, int out_size, void* d_ws, size_t ws_size,
                              hipStream_t stream) {
    (void)in_sizes; (void)n_in; (void)out_size; (void)ws_size;
    const float* x  = (const float*)d_in[0];
    const float* Ws = (const float*)d_in[1];
    const float* bs = (const float*)d_in[2];
    const float* We = (const float*)d_in[3];
    const float* be = (const float*)d_in[4];
    const float* Ww = (const float*)d_in[5];
    const float* bw = (const float*)d_in[6];
    const float* vv = (const float*)d_in[7];
    const float* Wi = (const float*)d_in[8];
    const float* bi = (const float*)d_in[9];
    const float* Wa = (const float*)d_in[10];
    const float* ba = (const float*)d_in[11];
    float* out = (float*)d_out;
    uint32_t* ws = (uint32_t*)d_ws;

    // 2048 + 16384 fragment words
    prep_weights<<<(18432 + 255) / 256, 256, 0, stream>>>(We, Ww, ws);
    groupnet_kernel<<<Bsz, 256, 0, stream>>>(x, Ws, bs, be, bw, vv,
                                             Wi, bi, Wa, ba, ws, out);
}
